// RobustInference_24979529793899
// MI455X (gfx1250) — matrix operand order repaired and checked by reading, not compiled, not run on hardware
//
#include <hip/hip_runtime.h>
#include <hip/hip_bf16.h>

// Problem constants (from reference setup_inputs)
#define B_BATCH 64
#define D_DIM   3072
#define N_CAND  8000
#define L_DIM   64
#define NC_CLS  10

typedef _Float16 v16h __attribute__((ext_vector_type(16)));
typedef _Float16 v8h  __attribute__((ext_vector_type(8)));
typedef float    v8f  __attribute__((ext_vector_type(8)));

struct HPair { v8h lo, hi; };
static __device__ __forceinline__ v16h mkv16(v8h lo, v8h hi) {
    HPair t{lo, hi};
    return __builtin_bit_cast(v16h, t);
}
static __device__ __forceinline__ v8h ldv8h(const _Float16* p) {
    return *(const v8h*)p;
}
// Fast sigmoid: v_exp_f32 + v_rcp_f32 (approximate rcp, ~1 ulp; avoids the
// full IEEE divide expansion that __frcp_rn / operator/ produce).
static __device__ __forceinline__ float fast_sigmoid(float v) {
    return __builtin_amdgcn_rcpf(1.0f + __expf(-v));
}

// ---------------------------------------------------------------------------
// Prep kernels
// ---------------------------------------------------------------------------

// minbuf[c*64+b] holds packed (monotonic loss bits << 32) | candidate idx
__global__ void prep_init(unsigned long long* minbuf) {
    int t = blockIdx.x * blockDim.x + threadIdx.x;
    if (t < NC_CLS * B_BATCH) minbuf[t] = ~0ull;
}

// dec_w (f32, [NC][L][D]) -> wt (f16, [NC][D][L])  : K-contiguous for WMMA B frags
__global__ void prep_w(const float* __restrict__ dec_w, _Float16* __restrict__ wt) {
    int t = blockIdx.x * blockDim.x + threadIdx.x;       // over NC*D
    if (t >= NC_CLS * D_DIM) return;
    int c = t / D_DIM, d = t % D_DIM;
    _Float16* o = wt + (size_t)t * L_DIM;
    #pragma unroll 8
    for (int l = 0; l < L_DIM; ++l)
        o[l] = (_Float16)dec_w[((size_t)(c * L_DIM + l)) * D_DIM + d];
}

// z (f32, [N][L]) -> zh f16 + kld[n] = 0.5*sum z^2
__global__ void prep_z(const float* __restrict__ z, _Float16* __restrict__ zh,
                       float* __restrict__ kld) {
    int n = blockIdx.x * blockDim.x + threadIdx.x;
    if (n >= N_CAND) return;
    float s = 0.f;
    #pragma unroll 8
    for (int l = 0; l < L_DIM; ++l) {
        float v = z[n * L_DIM + l];
        s += v * v;
        zh[n * L_DIM + l] = (_Float16)v;
    }
    kld[n] = 0.5f * s;
}

// x (f32, [B][D]) -> xh f16 + xsq[b] = ||x_b||^2
__global__ __launch_bounds__(256) void prep_x(const float* __restrict__ x,
                                              _Float16* __restrict__ xh,
                                              float* __restrict__ xsq) {
    int b = blockIdx.x;
    __shared__ float red[256];
    float s = 0.f;
    for (int d = threadIdx.x; d < D_DIM; d += 256) {
        float v = x[b * D_DIM + d];
        xh[b * D_DIM + d] = (_Float16)v;
        s += v * v;
    }
    red[threadIdx.x] = s;
    __syncthreads();
    for (int st = 128; st > 0; st >>= 1) {
        if (threadIdx.x < st) red[threadIdx.x] += red[threadIdx.x + st];
        __syncthreads();
    }
    if (threadIdx.x == 0) xsq[b] = red[0];
}

// ---------------------------------------------------------------------------
// Main fused kernel: decode -> r^2, cross GEMM -> packed argmin
// grid.x = ceil(N/128) n-tiles, grid.y = class.  256 thr = 8 waves x 16 cands.
// ---------------------------------------------------------------------------
__global__ __launch_bounds__(256) void robust_main(
    const _Float16* __restrict__ zh,   // [N][L]
    const _Float16* __restrict__ wt,   // [NC][D][L]
    const _Float16* __restrict__ xh,   // [B][D]
    const float*    __restrict__ dec_b,// [NC][D] f32
    const float*    __restrict__ kld,  // [N]
    const float*    __restrict__ xsq,  // [B]
    unsigned long long* __restrict__ minbuf) // [NC][B]
{
    __shared__ _Float16 stage[8][16][32];     // per-wave rec re-layout buffer
    __shared__ unsigned long long smin[B_BATCH];

    const int c    = blockIdx.y;
    const int wave = threadIdx.x >> 5;
    const int lane = threadIdx.x & 31;
    const int grp  = lane >> 4;               // lane group (0/1)
    const int lidx = lane & 15;
    const int n_base = blockIdx.x * 128 + wave * 16;

    if (threadIdx.x < B_BATCH) smin[threadIdx.x] = ~0ull;
    __syncthreads();

    // ---- A fragments for z rows n_base..n_base+15 (K = L = 64, two frags)
    // A 16x32 f16 layout: lane(grp,m=lidx), half h -> K = (h<8?0:16) + 8*grp + h%8
    v16h a0 = {}, a1 = {};
    {
        const int zrow = n_base + lidx;
        if (zrow < N_CAND) {
            const _Float16* zp = zh + (size_t)zrow * L_DIM + grp * 8;
            a0 = mkv16(ldv8h(zp),      ldv8h(zp + 16));
            a1 = mkv16(ldv8h(zp + 32), ldv8h(zp + 48));
        }
    }

    v8f acc0 = {}, acc1 = {}, acc2 = {}, acc3 = {};   // cross[16n x 64b]
    v8f rsq = {};                                      // per-lane partial r^2

    for (int d0 = 0; d0 < D_DIM; d0 += 32) {
        // ---- decode rec tile 16n x 32d via WMMA, sigmoid, stash to LDS
        #pragma unroll
        for (int dh = 0; dh < 2; ++dh) {
            const int dcol = d0 + dh * 16 + lidx;
            // B 32x16 f16 layout: lane col N=lidx, half h -> K = (h<8?0:16) + 8*grp + h%8
            const _Float16* wp = wt + ((size_t)c * D_DIM + dcol) * L_DIM + grp * 8;
            v16h blo = mkv16(ldv8h(wp),      ldv8h(wp + 16));   // l = (h<8?0:16)+grp*8+h%8
            v16h bhi = mkv16(ldv8h(wp + 32), ldv8h(wp + 48));   // l = 32+(h<8?0:16)+grp*8+h%8
            float bias = dec_b[c * D_DIM + dcol];
            v8f t;
            #pragma unroll
            for (int i = 0; i < 8; ++i) t[i] = bias;            // bias splat as C
            t = __builtin_amdgcn_wmma_f32_16x16x32_f16(false, a0, false, blo,
                                                       (short)0, t, false, false);
            t = __builtin_amdgcn_wmma_f32_16x16x32_f16(false, a1, false, bhi,
                                                       (short)0, t, false, false);
            #pragma unroll
            for (int i = 0; i < 8; ++i) {
                float s = fast_sigmoid(t[i]);
                rsq[i] += s * s;
                // C layout: row = i + 8*grp, col = lidx (within dh tile)
                stage[wave][i + 8 * grp][dh * 16 + lidx] = (_Float16)s;
            }
        }
        __builtin_amdgcn_wave_barrier();
        // ---- reload rec as A fragment (rows=n, K=32 d's)
        const _Float16* sp = &stage[wave][lidx][8 * grp];
        v16h arec = mkv16(ldv8h(sp), ldv8h(sp + 16));
        __builtin_amdgcn_wave_barrier();
        // ---- cross += rec(16x32) @ xT(32x64) : 4 column tiles of images
        #pragma unroll
        for (int t4 = 0; t4 < 4; ++t4) {
            const _Float16* xp = xh + (size_t)(t4 * 16 + lidx) * D_DIM + d0 + grp * 8;
            v16h bx = mkv16(ldv8h(xp), ldv8h(xp + 16));
            v8f& acc = (t4 == 0) ? acc0 : (t4 == 1) ? acc1 : (t4 == 2) ? acc2 : acc3;
            acc = __builtin_amdgcn_wmma_f32_16x16x32_f16(false, arec, false, bx,
                                                         (short)0, acc, false, false);
        }
    }

    // ---- reduce rsq across the 16 lanes of each group (butterfly)
    #pragma unroll
    for (int m = 1; m < 16; m <<= 1) {
        #pragma unroll
        for (int i = 0; i < 8; ++i) rsq[i] += __shfl_xor(rsq[i], m, 32);
    }
    // now rsq[i] = full ||rec_n||^2 for row n = n_base + i + 8*grp (all lanes in grp)

    float kldr[8];
    #pragma unroll
    for (int i = 0; i < 8; ++i) {
        int n_i = n_base + i + 8 * grp;
        kldr[i] = (n_i < N_CAND) ? kld[n_i] : 0.f;
    }

    #pragma unroll
    for (int t4 = 0; t4 < 4; ++t4) {
        const v8f& acc = (t4 == 0) ? acc0 : (t4 == 1) ? acc1 : (t4 == 2) ? acc2 : acc3;
        float xs = xsq[t4 * 16 + lidx];
        unsigned long long bk = ~0ull;
        #pragma unroll
        for (int i = 0; i < 8; ++i) {
            int n_i = n_base + i + 8 * grp;
            if (n_i < N_CAND) {
                float lv = xs + rsq[i] - 2.0f * acc[i] + kldr[i];
                unsigned u = __float_as_uint(lv);
                u ^= (u >> 31) ? 0xFFFFFFFFu : 0x80000000u;   // monotonic map
                unsigned long long k =
                    ((unsigned long long)u << 32) | (unsigned)n_i;
                bk = (k < bk) ? k : bk;
            }
        }
        // merge partner half-wave (rows 0-7 vs 8-15 of same column b)
        unsigned long long ok = __shfl_xor(bk, 16, 32);
        bk = (ok < bk) ? ok : bk;
        if (grp == 0) atomicMin(&smin[t4 * 16 + lidx], bk);
    }
    __syncthreads();
    if (threadIdx.x < B_BATCH)
        atomicMin(&minbuf[c * B_BATCH + threadIdx.x], smin[threadIdx.x]);
}

// ---------------------------------------------------------------------------
// Finalize: decode winners in fp32, emit logits/recs_sel/mus/logvars
// ---------------------------------------------------------------------------
__global__ __launch_bounds__(128) void robust_finalize(
    const float* __restrict__ x, const float* __restrict__ z,
    const float* __restrict__ dec_w, const float* __restrict__ dec_b,
    const unsigned long long* __restrict__ minbuf, float* __restrict__ out)
{
    constexpr size_t REC_OFF = NC_CLS * B_BATCH;                          // 640
    constexpr size_t MU_OFF  = REC_OFF + (size_t)NC_CLS * B_BATCH * D_DIM;
    constexpr size_t LV_OFF  = MU_OFF + (size_t)NC_CLS * B_BATCH * L_DIM;

    const int cb = blockIdx.x;
    const int c = cb / B_BATCH, b = cb % B_BATCH;

    __shared__ int sidx;
    __shared__ float zrow[L_DIM];
    __shared__ float sred[128];

    if (threadIdx.x == 0) {
        int idx = (int)(minbuf[cb] & 0xffffffffu);
        if (idx < 0 || idx >= N_CAND) idx = 0;
        sidx = idx;
    }
    __syncthreads();
    const int idx = sidx;
    if (threadIdx.x < L_DIM) zrow[threadIdx.x] = z[(size_t)idx * L_DIM + threadIdx.x];
    __syncthreads();

    float* recs = out + REC_OFF + (size_t)cb * D_DIM;
    float sse = 0.f;
    for (int d = threadIdx.x; d < D_DIM; d += 128) {
        float a = dec_b[c * D_DIM + d];
        #pragma unroll 8
        for (int l = 0; l < L_DIM; ++l)
            a = fmaf(zrow[l], dec_w[((size_t)(c * L_DIM + l)) * D_DIM + d], a);
        float r = fast_sigmoid(a);
        recs[d] = r;
        float df = x[b * D_DIM + d] - r;
        sse += df * df;
    }
    sred[threadIdx.x] = sse;
    __syncthreads();
    for (int s = 64; s > 0; s >>= 1) {
        if (threadIdx.x < s) sred[threadIdx.x] += sred[threadIdx.x + s];
        __syncthreads();
    }
    if (threadIdx.x < L_DIM) {
        out[MU_OFF + (size_t)cb * L_DIM + threadIdx.x] = zrow[threadIdx.x];
        out[LV_OFF + (size_t)cb * L_DIM + threadIdx.x] = 0.f;
    }
    if (threadIdx.x == 0) {
        float kld = 0.f;
        #pragma unroll 8
        for (int l = 0; l < L_DIM; ++l) kld += zrow[l] * zrow[l];
        out[b * NC_CLS + c] = -(sred[0] + 0.5f * kld);   // logits (B, NC)
    }
}

// ---------------------------------------------------------------------------
extern "C" void kernel_launch(void* const* d_in, const int* in_sizes, int n_in,
                              void* d_out, int out_size, void* d_ws, size_t ws_size,
                              hipStream_t stream) {
    const float* x     = (const float*)d_in[0];
    const float* z     = (const float*)d_in[1];
    const float* dec_w = (const float*)d_in[2];
    const float* dec_b = (const float*)d_in[3];
    float* out = (float*)d_out;
    (void)in_sizes; (void)n_in; (void)out_size; (void)ws_size;

    // Carve workspace (256B-aligned slabs, ~5.4 MB total)
    char* ws = (char*)d_ws;
    size_t off = 0;
    auto carve = [&](size_t bytes) -> void* {
        void* p = ws + off;
        off += (bytes + 255) & ~(size_t)255;
        return p;
    };
    _Float16* wt  = (_Float16*)carve((size_t)NC_CLS * D_DIM * L_DIM * 2); // W^T f16
    _Float16* zh  = (_Float16*)carve((size_t)N_CAND * L_DIM * 2);
    _Float16* xh  = (_Float16*)carve((size_t)B_BATCH * D_DIM * 2);
    float*    kld = (float*)carve((size_t)N_CAND * 4);
    float*    xsq = (float*)carve((size_t)B_BATCH * 4);
    unsigned long long* minbuf =
        (unsigned long long*)carve((size_t)NC_CLS * B_BATCH * 8);

    prep_init<<<3, 256, 0, stream>>>(minbuf);
    prep_w<<<(NC_CLS * D_DIM + 255) / 256, 256, 0, stream>>>(dec_w, wt);
    prep_z<<<(N_CAND + 255) / 256, 256, 0, stream>>>(z, zh, kld);
    prep_x<<<B_BATCH, 256, 0, stream>>>(x, xh, xsq);

    dim3 grid((N_CAND + 127) / 128, NC_CLS);
    robust_main<<<grid, 256, 0, stream>>>(zh, wt, xh, dec_b, kld, xsq, minbuf);

    robust_finalize<<<NC_CLS * B_BATCH, 128, 0, stream>>>(x, z, dec_w, dec_b,
                                                          minbuf, out);
}
